// Pointnet_Features_35081292874355
// MI455X (gfx1250) — matrix operand order repaired and checked by reading, not compiled, not run on hardware
//
#include <hip/hip_runtime.h>
#include <hip/hip_bf16.h>
#include <math.h>

typedef __attribute__((ext_vector_type(16))) _Float16 v16h;
typedef __attribute__((ext_vector_type(8)))  float    v8f;

#define NPTS  2048
#define BATCH 8
#define EPSLN 1e-5f

__device__ __forceinline__ float sigmoidf_(float x) { return 1.0f / (1.0f + __expf(-x)); }

// ---------------------------------------------------------------------------
// A fragment (16x32 f16): lane = M row (lane&15); lanes 0-15 K 0..15,
// lanes 16-31 K 16..31. Two contiguous 8-float runs -> 4x float4.
// Requires K % 32 == 0.
// ---------------------------------------------------------------------------
__device__ __forceinline__ v16h load_a_frag(const float* __restrict__ Wrow, int k0, int hi)
{
    const float4* p0 = (const float4*)(Wrow + k0 + hi * 16);
    const float4* p1 = (const float4*)(Wrow + k0 + 8 + hi * 16);
    float4 q0 = p0[0], q1 = p0[1], q2 = p1[0], q3 = p1[1];
    v16h a;
    a[0] = (_Float16)q0.x; a[1] = (_Float16)q0.y; a[2] = (_Float16)q0.z; a[3] = (_Float16)q0.w;
    a[4] = (_Float16)q1.x; a[5] = (_Float16)q1.y; a[6] = (_Float16)q1.z; a[7] = (_Float16)q1.w;
    a[8] = (_Float16)q2.x; a[9] = (_Float16)q2.y; a[10] = (_Float16)q2.z; a[11] = (_Float16)q2.w;
    a[12] = (_Float16)q3.x; a[13] = (_Float16)q3.y; a[14] = (_Float16)q3.z; a[15] = (_Float16)q3.w;
    return a;
}

// ---------------------------------------------------------------------------
// B fragment (32x16 f16): lane = column n (lane&15); lanes 0-15 K=0..15,
// lanes 16-31 K=16..31. 16 loads at compile-time-constant offsets (N = NPTS),
// coalesced across the half-wave.
// ---------------------------------------------------------------------------
__device__ __forceinline__ v16h load_b_frag(const float* __restrict__ Xcol, int k0, int hi)
{
    const float* p = Xcol + (long)(k0 + hi * 16) * NPTS;
    v16h bf;
#pragma unroll
    for (int e = 0; e < 16; ++e) bf[e] = (_Float16)p[(long)e * NPTS];
    return bf;
}

// ---------------------------------------------------------------------------
// Batched GEMM via v_wmma_f32_16x16x32_f16, wave tile = (MT*16) x (NT*16).
//   Y[b,m,n] = sum_k W[m,k] * X[b*xBatchStride + k*NPTS + n]  (+ epilogue)
// mode 0: + bias (may be null) | mode 1: + bias, BN, ReLU | mode 2: + bias, + resid
// Requires K % 32 == 0, M % (16*MT) == 0.
// ---------------------------------------------------------------------------
template<int MT, int NT>
__global__ __launch_bounds__(256)
void gemm_wmma_kernel(const float* __restrict__ W,
                      const float* __restrict__ X,
                      float* __restrict__ Y,
                      const float* __restrict__ bias,
                      const float* __restrict__ bng,
                      const float* __restrict__ bnb,
                      const float* __restrict__ bnm,
                      const float* __restrict__ bnv,
                      const float* __restrict__ resid,
                      int M, int K, long xBatchStride, int mode)
{
    const int lane = threadIdx.x & 31;
    const int wave = threadIdx.x >> 5;
    const int tilesM = M / (16 * MT);
    const int tilesN = NPTS / (16 * NT);
    const long tilesPerBatch = (long)tilesM * tilesN;
    const long totalTiles    = tilesPerBatch * BATCH;

    long tile = (long)blockIdx.x * 8 + wave;
    if (tile >= totalTiles) return;

    const int b  = (int)(tile / tilesPerBatch);
    const int r  = (int)(tile % tilesPerBatch);
    const int mt = r / tilesN;
    const int nt = r % tilesN;
    const int m0 = mt * 16 * MT;
    const int n0 = nt * 16 * NT;

    const int hl = lane & 15;   // 0..15
    const int hi = lane >> 4;   // 0 or 1

    const float* Xb = X + (long)b * xBatchStride;

    const float* Wrow[MT];
#pragma unroll
    for (int mi = 0; mi < MT; ++mi)
        Wrow[mi] = W + (long)(m0 + mi * 16 + hl) * K;
    const float* Xcol[NT];
#pragma unroll
    for (int ni = 0; ni < NT; ++ni)
        Xcol[ni] = Xb + (n0 + ni * 16 + hl);

    v8f acc[MT][NT];
#pragma unroll
    for (int mi = 0; mi < MT; ++mi)
#pragma unroll
        for (int ni = 0; ni < NT; ++ni)
            acc[mi][ni] = (v8f){};

    for (int k0 = 0; k0 < K; k0 += 32) {
        // Prefetch the K tile after next of the strided activation stream (and weights):
        // emits global_prefetch_b8; speculative, dropped if out of range.
        if (k0 + 64 < K) {
#pragma unroll
            for (int ni = 0; ni < NT; ++ni)
                __builtin_prefetch(Xcol[ni] + (long)(k0 + 64 + hi * 16) * NPTS, 0, 1);
#pragma unroll
            for (int mi = 0; mi < MT; ++mi)
                __builtin_prefetch(Wrow[mi] + k0 + 64, 0, 1);
        }
        v16h bf[NT];
#pragma unroll
        for (int ni = 0; ni < NT; ++ni) bf[ni] = load_b_frag(Xcol[ni], k0, hi);
        v16h af[MT];
#pragma unroll
        for (int mi = 0; mi < MT; ++mi) af[mi] = load_a_frag(Wrow[mi], k0, hi);
#pragma unroll
        for (int mi = 0; mi < MT; ++mi)
#pragma unroll
            for (int ni = 0; ni < NT; ++ni)
                acc[mi][ni] = __builtin_amdgcn_wmma_f32_16x16x32_f16(
                    false, af[mi], false, bf[ni], (short)0, acc[mi][ni], false, false);
    }

    // C/D layout: vgpr rr -> row (+ rr + hi*8), column hl
#pragma unroll
    for (int mi = 0; mi < MT; ++mi) {
#pragma unroll
        for (int rr = 0; rr < 8; ++rr) {
            int m = m0 + mi * 16 + rr + hi * 8;
            float bb = bias ? bias[m] : 0.0f;
            float bn_scale = 1.0f, bn_off = 0.0f;
            if (mode == 1) {
                bn_scale = __frsqrt_rn(bnv[m] + EPSLN) * bng[m];
                bn_off   = bnb[m] - bnm[m] * bn_scale;
            }
#pragma unroll
            for (int ni = 0; ni < NT; ++ni) {
                int n = n0 + ni * 16 + hl;
                float v = acc[mi][ni][rr] + bb;
                if (mode == 1) {
                    v = fmaxf(v * bn_scale + bn_off, 0.0f);
                } else if (mode == 2) {
                    v += resid[((long)b * M + m) * NPTS + n];
                }
                Y[((long)b * M + m) * NPTS + n] = v;
            }
        }
    }
}

// ---------------------------------------------------------------------------
// Stage 1 fused: transpose(points) -> linear(K=3) -> BN -> ReLU.
// ---------------------------------------------------------------------------
__global__ __launch_bounds__(256)
void stage1_fused_kernel(const float* __restrict__ pts,  // (B, N, 3)
                         const float* __restrict__ w,    // (64, 3)
                         const float* __restrict__ bias,
                         const float* __restrict__ bng,
                         const float* __restrict__ bnb,
                         const float* __restrict__ bnm,
                         const float* __restrict__ bnv,
                         float* __restrict__ y)          // (B, 64, N)
{
    long idx = (long)blockIdx.x * blockDim.x + threadIdx.x;
    long total = (long)BATCH * 64 * NPTS;
    if (idx >= total) return;
    int  n  = (int)(idx % NPTS);
    long bm = idx / NPTS;
    int  m  = (int)(bm % 64);
    int  b  = (int)(bm / 64);

    const float* pp = pts + ((long)b * NPTS + n) * 3;
    float v = w[m * 3 + 0] * pp[0] + w[m * 3 + 1] * pp[1] + w[m * 3 + 2] * pp[2] + bias[m];
    float sc = __frsqrt_rn(bnv[m] + EPSLN) * bng[m];
    v = fmaxf((v - bnm[m]) * sc + bnb[m], 0.0f);
    y[((long)b * 64 + m) * NPTS + n] = v;
}

// ---------------------------------------------------------------------------
// Concatenate Bw (16,di) and Cw (16,di) into WBC (32,di).
// ---------------------------------------------------------------------------
__global__ __launch_bounds__(256)
void concat_bc_kernel(const float* __restrict__ Bw, const float* __restrict__ Cw,
                      float* __restrict__ WBC, int di)
{
    int idx = blockIdx.x * blockDim.x + threadIdx.x;
    int total = 16 * di;
    if (idx >= total) return;
    WBC[idx] = Bw[idx];
    WBC[total + idx] = Cw[idx];
}

// ---------------------------------------------------------------------------
// Causal depthwise conv (width 4, pad 3 left) + bias + SiLU.
// ---------------------------------------------------------------------------
__global__ __launch_bounds__(256)
void conv_silu_kernel(const float* __restrict__ xp,   // (B, 2di, N), conv reads rows 0..di-1
                      const float* __restrict__ cw,   // (di, 4)
                      const float* __restrict__ cb,   // (di)
                      float* __restrict__ xc,         // (B, di, N)
                      int di)
{
    long idx = (long)blockIdx.x * blockDim.x + threadIdx.x;
    long total = (long)BATCH * di * NPTS;
    if (idx >= total) return;
    int  n  = (int)(idx % NPTS);
    long bd = idx / NPTS;
    int  d  = (int)(bd % di);
    int  b  = (int)(bd / di);

    const float* row = xp + ((long)b * 2 * di + d) * NPTS;
    const float* w   = cw + (long)d * 4;
    float acc = 0.0f;
#pragma unroll
    for (int j = 0; j < 4; ++j) {
        int t = n - 3 + j;
        if (t >= 0) acc += w[j] * row[t];
    }
    acc += cb[d];
    acc = acc * sigmoidf_(acc);
    xc[((long)b * di + d) * NPTS + n] = acc;
}

// ---------------------------------------------------------------------------
// Selective scan. One wave = 2 channels x 16 states; t unrolled by 4 with
// float4 loads. Bs/Cs come from the combined (B, 32, N) buffer (rows 0-15 = B,
// rows 16-31 = C). y is written into xp's dead xc rows.
// ---------------------------------------------------------------------------
__global__ __launch_bounds__(256)
void ssm_scan_kernel(const float* __restrict__ xp,   // (B, 2di, N): z rows at di+d
                     const float* __restrict__ xc,   // (B, di, N)
                     const float* __restrict__ BC,   // (B, 32, N)
                     const float* __restrict__ A,    // (di, 16)
                     const float* __restrict__ Dp,   // (di)
                     float* __restrict__ ybuf,       // == xp (rows d)
                     int di)
{
    int wave  = (int)(((long)blockIdx.x * blockDim.x + threadIdx.x) >> 5);
    int lane  = threadIdx.x & 31;
    int s     = lane & 15;
    int cpair = lane >> 4;
    int wavesPerBatch = di >> 1;
    int b = wave / wavesPerBatch;
    if (b >= BATCH) return;
    int d = (wave % wavesPerBatch) * 2 + cpair;

    const float a  = -__expf(A[(long)d * 16 + s]);
    const float Dd = Dp[d];
    const float* zrow = xp + ((long)b * 2 * di + di + d) * NPTS;
    const float* xrow = xc + ((long)b * di + d) * NPTS;
    const float* Brow = BC + ((long)b * 32 + s) * NPTS;
    const float* Crow = BC + ((long)b * 32 + 16 + s) * NPTS;
    float*       yrow = ybuf + ((long)b * 2 * di + d) * NPTS;

    float h = 0.0f;
    for (int t = 0; t < NPTS; t += 4) {
        float4 z4 = *(const float4*)(zrow + t);
        float4 B4 = *(const float4*)(Brow + t);
        float4 C4 = *(const float4*)(Crow + t);
        float4 x4 = *(const float4*)(xrow + t);
        float y4[4];
        const float zz[4] = {z4.x, z4.y, z4.z, z4.w};
        const float BB[4] = {B4.x, B4.y, B4.z, B4.w};
        const float CC[4] = {C4.x, C4.y, C4.z, C4.w};
        const float xx[4] = {x4.x, x4.y, x4.z, x4.w};
#pragma unroll
        for (int j = 0; j < 4; ++j) {
            h = a * h + sigmoidf_(zz[j]) * BB[j];
            float part = h * CC[j];
            part += __shfl_xor(part, 1, 16);
            part += __shfl_xor(part, 2, 16);
            part += __shfl_xor(part, 4, 16);
            part += __shfl_xor(part, 8, 16);
            y4[j] = part + Dd * xx[j];
        }
        if (s == 0) *(float4*)(yrow + t) = make_float4(y4[0], y4[1], y4[2], y4[3]);
    }
}

// ---------------------------------------------------------------------------
// LayerNorm over channels, per point. One block per (b, n). In-place capable.
// ---------------------------------------------------------------------------
__global__ __launch_bounds__(256)
void layernorm_kernel(const float* __restrict__ o,   // (B, C, N)
                      const float* __restrict__ g,
                      const float* __restrict__ bt,
                      float* __restrict__ outx,      // may == o
                      int C)
{
    int n = blockIdx.x % NPTS;
    int b = blockIdx.x / NPTS;
    int tid = threadIdx.x;
    __shared__ float red[256];

    float vals[8];
    int cnt = 0;
    float sum = 0.0f;
    for (int c = tid; c < C; c += 256) {
        float v = o[((long)b * C + c) * NPTS + n];
        vals[cnt++] = v;
        sum += v;
    }
    red[tid] = sum; __syncthreads();
    for (int off = 128; off > 0; off >>= 1) {
        if (tid < off) red[tid] += red[tid + off];
        __syncthreads();
    }
    float mu = red[0] / (float)C;
    __syncthreads();

    float vsum = 0.0f; cnt = 0;
    for (int c = tid; c < C; c += 256) { float dv = vals[cnt++] - mu; vsum += dv * dv; }
    red[tid] = vsum; __syncthreads();
    for (int off = 128; off > 0; off >>= 1) {
        if (tid < off) red[tid] += red[tid + off];
        __syncthreads();
    }
    float rstd = __frsqrt_rn(red[0] / (float)C + EPSLN);

    cnt = 0;
    for (int c = tid; c < C; c += 256) {
        float v = (vals[cnt++] - mu) * rstd * g[c] + bt[c];
        outx[((long)b * C + c) * NPTS + n] = v;
    }
}

// ---------------------------------------------------------------------------
// Max over N. One block per (b, c).
// ---------------------------------------------------------------------------
__global__ __launch_bounds__(256)
void max_reduce_kernel(const float* __restrict__ x, float* __restrict__ out, int C)
{
    int c = blockIdx.x % C;
    int b = blockIdx.x / C;
    int tid = threadIdx.x;
    __shared__ float red[256];
    const float* row = x + ((long)b * C + c) * NPTS;
    float m = -3.402823466e38f;
    for (int n = tid; n < NPTS; n += 256) m = fmaxf(m, row[n]);
    red[tid] = m; __syncthreads();
    for (int off = 128; off > 0; off >>= 1) {
        if (tid < off) red[tid] = fmaxf(red[tid], red[tid + off]);
        __syncthreads();
    }
    if (tid == 0) out[(long)b * C + c] = red[0];
}

// ---------------------------------------------------------------------------
extern "C" void kernel_launch(void* const* d_in, const int* in_sizes, int n_in,
                              void* d_out, int out_size, void* d_ws, size_t ws_size,
                              hipStream_t stream)
{
    (void)in_sizes; (void)n_in; (void)out_size; (void)ws_size;

    const int dims_out[3] = {64, 128, 1024};

    // Workspace layout (f32 elements), worst case = stage 3 (~514 MB).
    float* Abuf = (float*)d_ws;                          // (B, 1024, N) out_proj + LN output
    size_t fAct = (size_t)BATCH * 1024 * NPTS;
    float* Bbuf = Abuf + fAct;                           // (B, 1024, N) post-linear x1 (residual)
    float* XP   = Bbuf + fAct;                           // (B, 4096, N) in_proj out; y reuses xc rows
    size_t fXP  = (size_t)BATCH * 4096 * NPTS;
    float* XC   = XP + fXP;                              // (B, 2048, N) conv output
    size_t fXC  = (size_t)BATCH * 2048 * NPTS;
    float* BC   = XC + fXC;                              // (B, 32, N) combined Bs/Cs
    size_t fBC  = (size_t)BATCH * 32 * NPTS;
    float* WBC  = BC + fBC;                              // (32, 2048) concatenated Bw/Cw weights

    const float* points = (const float*)d_in[0];

    for (int st = 0; st < 3; ++st) {
        const int cout = dims_out[st], di = 2 * cout;
        const float* const* p = (const float* const*)(d_in + 1 + st * 18);
        const float* w    = p[0];
        const float* bsv  = p[1];
        const float* bng  = p[2];
        const float* bnb  = p[3];
        const float* bnm  = p[4];
        const float* bnv  = p[5];
        const float* in_w = p[6];
        const float* in_b = p[7];
        const float* cw   = p[8];
        const float* cb   = p[9];
        const float* Ap   = p[10];
        const float* Dp   = p[11];
        const float* Bw   = p[12];
        const float* Cw   = p[13];
        const float* ow   = p[14];
        const float* ob   = p[15];
        const float* lng  = p[16];
        const float* lnb  = p[17];

        // 1) stage linear + BN + ReLU -> Bbuf
        if (st == 0) {
            long total = (long)BATCH * 64 * NPTS;
            stage1_fused_kernel<<<(int)((total + 255) / 256), 256, 0, stream>>>(
                points, w, bsv, bng, bnb, bnm, bnv, Bbuf);
        } else {
            const int cin = dims_out[st - 1];
            long tiles  = (long)BATCH * (cout / 64) * (NPTS / 32);
            gemm_wmma_kernel<4, 2><<<(int)((tiles + 7) / 8), 256, 0, stream>>>(
                w, Abuf, Bbuf, bsv, bng, bnb, bnm, bnv, nullptr,
                cout, cin, (long)cin * NPTS, 1);
        }

        // 2) in_proj: Bbuf -> XP (B, 2di, N)
        {
            long tiles = (long)BATCH * ((2 * di) / 64) * (NPTS / 32);
            gemm_wmma_kernel<4, 2><<<(int)((tiles + 7) / 8), 256, 0, stream>>>(
                in_w, Bbuf, XP, in_b, nullptr, nullptr, nullptr, nullptr, nullptr,
                2 * di, cout, (long)cout * NPTS, 0);
        }

        // 3) causal depthwise conv + SiLU: XP(xc rows) -> XC
        {
            long total = (long)BATCH * di * NPTS;
            conv_silu_kernel<<<(int)((total + 255) / 256), 256, 0, stream>>>(XP, cw, cb, XC, di);
        }

        // 4) merged Bs/Cs projection (M=32): XC -> BC, single pass over XC
        {
            concat_bc_kernel<<<(16 * di + 255) / 256, 256, 0, stream>>>(Bw, Cw, WBC, di);
            long tiles = (long)BATCH * 1 * (NPTS / 32);
            gemm_wmma_kernel<2, 2><<<(int)((tiles + 7) / 8), 256, 0, stream>>>(
                WBC, XC, BC, nullptr, nullptr, nullptr, nullptr, nullptr, nullptr,
                32, di, (long)di * NPTS, 0);
        }

        // 5) selective scan: y written into XP's xc rows
        {
            long waves   = (long)BATCH * (di / 2);
            long threads = waves * 32;
            ssm_scan_kernel<<<(int)((threads + 255) / 256), 256, 0, stream>>>(
                XP, XC, BC, Ap, Dp, XP, di);
        }

        // 6) out_proj + residual: y (XP rows 0..di-1, batch stride 2di*N) -> Abuf
        {
            long tiles = (long)BATCH * (cout / 64) * (NPTS / 32);
            gemm_wmma_kernel<4, 2><<<(int)((tiles + 7) / 8), 256, 0, stream>>>(
                ow, XP, Abuf, ob, nullptr, nullptr, nullptr, nullptr, Bbuf,
                cout, di, (long)(2 * di) * NPTS, 2);
        }

        // 7) LayerNorm over channels (in place on Abuf)
        layernorm_kernel<<<BATCH * NPTS, 256, 0, stream>>>(Abuf, lng, lnb, Abuf, cout);
    }

    // Final max over N: (B, 1024)
    max_reduce_kernel<<<BATCH * 1024, 256, 0, stream>>>(Abuf, (float*)d_out, 1024);
}
